// CustomAttention_17506286698773
// MI455X (gfx1250) — matrix operand order repaired and checked by reading, not compiled, not run on hardware
//
#include <hip/hip_runtime.h>

// ---------------------------------------------------------------------------
// Flash-attention for MI455X (gfx1250, wave32, WMMA f16 16x16x32, f32 acc).
// B=4, H=16, S=2048, d_k=64. One workgroup = (b, h, 128-row query tile),
// 8 waves, each wave owns a 16-row query strip. K/V streamed in 64-key
// chunks through LDS with register double-buffering and online softmax.
// 16 v_wmma per chunk per wave (8 QK^T + 8 PV).
// ---------------------------------------------------------------------------

#define S_LEN   2048
#define DM      1024
#define NH      16
#define DK      64
#define QT      16            // query rows per wave
#define NW      8             // waves per workgroup
#define WGQ     (QT * NW)     // 128 query rows per workgroup
#define KC      64            // keys per chunk

typedef __attribute__((ext_vector_type(16))) _Float16 v16h;
typedef __attribute__((ext_vector_type(8)))  _Float16 v8h;
typedef __attribute__((ext_vector_type(2)))  _Float16 v2h;
typedef __attribute__((ext_vector_type(8)))  float    v8f;

// LDS row stride in halfs: multiple of 8 (16B-aligned b128 rows); 72*2=144 B
// rows -> row*36 mod 64 banks, distinct for any 16 consecutive rows.
#define LSTR 72

__global__ __launch_bounds__(256)
void fa_fwd_kernel(const float* __restrict__ Q,
                   const float* __restrict__ K,
                   const float* __restrict__ V,
                   const int*   __restrict__ MSK,
                   float*       __restrict__ O)
{
    __shared__ __align__(16) _Float16 lds_q[NW][QT][LSTR];
    __shared__ __align__(16) _Float16 lds_k[KC][LSTR];      // [key][dim]
    __shared__ __align__(16) _Float16 lds_vt[DK][LSTR];     // [dim][key]
    __shared__ __align__(16) _Float16 lds_p[NW][QT][LSTR];

    const int tid  = threadIdx.x;
    const int wave = tid >> 5;
    const int lane = tid & 31;
    const int hi   = lane >> 4;     // half-wave select (0/1)
    const int ln   = lane & 15;

    const int blk   = blockIdx.x;       // 0..1023
    const int qtile = blk & 15;         // 16 query tiles of 128 rows
    const int h     = (blk >> 4) & (NH - 1);
    const int b     = blk >> 8;

    const int    q0      = qtile * WGQ + wave * QT;
    const size_t headOff = (size_t)h * DK;
    const float  scale   = 0.125f;      // 1/sqrt(64), folded into Q

    // ---- stage this wave's 16x64 Q strip into LDS (scaled, f32->f16) ----
    {
        const float* qb = Q + ((size_t)b * S_LEN + q0) * DM + headOff;
        #pragma unroll
        for (int i = 0; i < 32; ++i) {
            int e = lane + i * 32;              // 0..1023
            int r = e >> 6, c = e & 63;
            lds_q[wave][r][c] = (_Float16)(qb[(size_t)r * DM + c] * scale);
        }
    }

    // ---- A-fragments of Q (16x32 f16, two K-splits of d_k=64), loop-invariant ----
    v16h aq[2];
    #pragma unroll
    for (int c = 0; c < 2; ++c) {
        const _Float16* base = &lds_q[wave][ln][c * 32 + hi * 8];
        ((v8h*)&aq[c])[0] = *(const v8h*)(base);        // K = hi*8 + 0..7
        ((v8h*)&aq[c])[1] = *(const v8h*)(base + 16);   // K = 16 + hi*8 + 0..7
    }

    // ---- accumulators: O = 16x64 f32 (4 C-tiles), online-softmax stats ----
    v8f   oacc[4] = {};
    float mrow[8], lrow[8];
    #pragma unroll
    for (int r = 0; r < 8; ++r) { mrow[r] = -3.0e38f; lrow[r] = 0.0f; }

    const float* kb = K + (size_t)b * S_LEN * DM + headOff;
    const float* vb = V + (size_t)b * S_LEN * DM + headOff;
    const int*   mb = MSK + ((size_t)b * S_LEN + q0) * S_LEN;   // mask[b][q0+m][k]

    // ---- register prefetch buffers: per thread 8 keys (key = wave*8+i),
    //      dims {lane, lane+32}, converted to f16 at load ----
    v2h kpk[8], vpk[8];

    // prologue: load chunk 0
    #pragma unroll
    for (int i = 0; i < 8; ++i) {
        const int    key  = wave * 8 + i;
        const float* krow = kb + (size_t)key * DM;
        const float* vrow = vb + (size_t)key * DM;
        v2h kt; kt.x = (_Float16)krow[lane]; kt.y = (_Float16)krow[lane + 32];
        v2h vt; vt.x = (_Float16)vrow[lane]; vt.y = (_Float16)vrow[lane + 32];
        kpk[i] = kt; vpk[i] = vt;
    }

    for (int k0 = 0; k0 < S_LEN; k0 += KC) {
        __syncthreads();    // previous chunk's LDS fully consumed

        // ---- publish register-buffered K/V chunk into LDS ----
        #pragma unroll
        for (int i = 0; i < 8; ++i) {
            const int key = wave * 8 + i;
            lds_k [key][lane]      = kpk[i].x;
            lds_k [key][lane + 32] = kpk[i].y;
            lds_vt[lane][key]      = vpk[i].x;
            lds_vt[lane + 32][key] = vpk[i].y;
        }
        __syncthreads();

        // ---- issue next chunk's global loads early (hide under compute) ----
        if (k0 + KC < S_LEN) {
            #pragma unroll
            for (int i = 0; i < 8; ++i) {
                const int    key  = wave * 8 + i;
                const float* krow = kb + (size_t)(k0 + KC + key) * DM;
                const float* vrow = vb + (size_t)(k0 + KC + key) * DM;
                v2h kt; kt.x = (_Float16)krow[lane]; kt.y = (_Float16)krow[lane + 32];
                v2h vt; vt.x = (_Float16)vrow[lane]; vt.y = (_Float16)vrow[lane + 32];
                kpk[i] = kt; vpk[i] = vt;
            }
        }

        // ---- scores: S = Q(16x64) @ K^T(64x64) -> four 16x16 f32 tiles ----
        v8f s[4] = {};
        #pragma unroll
        for (int c = 0; c < 2; ++c) {
            #pragma unroll
            for (int j = 0; j < 4; ++j) {
                v16h bk;
                const _Float16* p = &lds_k[j * 16 + ln][c * 32 + hi * 8];
                ((v8h*)&bk)[0] = *(const v8h*)(p);
                ((v8h*)&bk)[1] = *(const v8h*)(p + 16);
                s[j] = __builtin_amdgcn_wmma_f32_16x16x32_f16(false, aq[c], false, bk,
                                                              (short)0, s[j],
                                                              false, false);
            }
        }

        // ---- mask + online softmax (C-layout: row = r + 8*hi, col = ln + j*16) ----
        float pv[4][8], rmax[8];
        #pragma unroll
        for (int r = 0; r < 8; ++r) {
            const int  qm = r + 8 * hi;
            const int* mr = mb + (size_t)qm * S_LEN + k0;
            float mx = -3.0e38f;
            #pragma unroll
            for (int j = 0; j < 4; ++j) {
                float x = mr[ln + j * 16] ? s[j][r] : -1.0e9f;
                pv[j][r] = x;
                mx = fmaxf(mx, x);
            }
            rmax[r] = mx;
        }
        #pragma unroll
        for (int d = 1; d < 16; d <<= 1)
            #pragma unroll
            for (int r = 0; r < 8; ++r)
                rmax[r] = fmaxf(rmax[r], __shfl_xor(rmax[r], d, 32));

        float corr[8], rsum[8];
        #pragma unroll
        for (int r = 0; r < 8; ++r) {
            float mn = fmaxf(mrow[r], rmax[r]);
            corr[r]  = __expf(mrow[r] - mn);
            mrow[r]  = mn;
            float sum = 0.0f;
            #pragma unroll
            for (int j = 0; j < 4; ++j) {
                pv[j][r] = __expf(pv[j][r] - mn);
                sum += pv[j][r];
            }
            rsum[r] = sum;
        }
        #pragma unroll
        for (int d = 1; d < 16; d <<= 1)
            #pragma unroll
            for (int r = 0; r < 8; ++r)
                rsum[r] += __shfl_xor(rsum[r], d, 32);
        #pragma unroll
        for (int r = 0; r < 8; ++r)
            lrow[r] = lrow[r] * corr[r] + rsum[r];

        // rescale running output before accumulating this chunk
        #pragma unroll
        for (int t = 0; t < 4; ++t)
            #pragma unroll
            for (int r = 0; r < 8; ++r)
                oacc[t][r] *= corr[r];

        // ---- stage P (f16) through per-wave LDS: C-layout -> A-layout ----
        #pragma unroll
        for (int r = 0; r < 8; ++r) {
            const int qm = r + 8 * hi;
            #pragma unroll
            for (int j = 0; j < 4; ++j)
                lds_p[wave][qm][ln + j * 16] = (_Float16)pv[j][r];
        }
        v16h ap[2];
        #pragma unroll
        for (int c = 0; c < 2; ++c) {
            const _Float16* pb = &lds_p[wave][ln][c * 32 + hi * 8];
            ((v8h*)&ap[c])[0] = *(const v8h*)(pb);          // K = hi*8 + 0..7
            ((v8h*)&ap[c])[1] = *(const v8h*)(pb + 16);     // K = 16 + hi*8 + 0..7
        }

        // ---- O += P(16x64) @ V(64x64): four 16x16 output tiles, 2 key-splits ----
        #pragma unroll
        for (int c = 0; c < 2; ++c) {
            #pragma unroll
            for (int t = 0; t < 4; ++t) {
                v16h bv;
                const _Float16* vp = &lds_vt[t * 16 + ln][c * 32 + hi * 8];
                ((v8h*)&bv)[0] = *(const v8h*)(vp);
                ((v8h*)&bv)[1] = *(const v8h*)(vp + 16);
                oacc[t] = __builtin_amdgcn_wmma_f32_16x16x32_f16(false, ap[c], false, bv,
                                                                 (short)0, oacc[t],
                                                                 false, false);
            }
        }
    }

    // ---- epilogue: O / l, coalesced f32 stores ----
    float* ob = O + ((size_t)b * S_LEN + q0) * DM + headOff;
    #pragma unroll
    for (int r = 0; r < 8; ++r) {
        const int   qm  = r + 8 * hi;
        const float inv = 1.0f / lrow[r];
        #pragma unroll
        for (int t = 0; t < 4; ++t)
            ob[(size_t)qm * DM + t * 16 + ln] = oacc[t][r] * inv;
    }
}

extern "C" void kernel_launch(void* const* d_in, const int* in_sizes, int n_in,
                              void* d_out, int out_size, void* d_ws, size_t ws_size,
                              hipStream_t stream)
{
    const float* Q = (const float*)d_in[0];
    const float* K = (const float*)d_in[1];
    const float* V = (const float*)d_in[2];
    const int*   M = (const int*)d_in[3];
    float*       O = (float*)d_out;
    (void)in_sizes; (void)n_in; (void)out_size; (void)d_ws; (void)ws_size;

    const int B = 4;
    dim3 grid(B * NH * (S_LEN / WGQ));   // 4 * 16 * 16 = 1024 workgroups
    dim3 block(256);                     // 8 waves (wave32)
    fa_fwd_kernel<<<grid, block, 0, stream>>>(Q, K, V, M, O);
}
